// TorchGNN_85650237817340
// MI455X (gfx1250) — matrix operand order repaired and checked by reading, not compiled, not run on hardware
//
#include <hip/hip_runtime.h>
#include <hip/hip_bf16.h>
#include <math.h>

// ---------------------------------------------------------------------------
// GNN propagation for MI455X (gfx1250), bf16 WMMA path.
// state: N x 64 fp32 (+ bf16 mirror for gathers)
// per prop: edge MLP (136->64->64->64) via v_wmma_f32_16x16x32_bf16,
//           atomic segment-sum, GRU via WMMA.
// ---------------------------------------------------------------------------

typedef __attribute__((ext_vector_type(16))) __bf16 v16bf;
typedef __attribute__((ext_vector_type(8)))  __bf16 v8bf;
typedef __attribute__((ext_vector_type(8)))  float  v8f;

#define HDIM   64
#define NNODE  50000
#define NEDGE  800000
#define ETILES (NEDGE / 16)   // 50000
#define NTILES (NNODE / 16)   // 3125
#define LDSR   72             // padded LDS row stride (bf16 elems), 144B = 9*16B

static __device__ __forceinline__ v8f wmma_acc(v16bf a, v16bf b, v8f c) {
  return __builtin_amdgcn_wmma_f32_16x16x32_bf16(
      /*neg_a=*/false, a, /*neg_b=*/false, b,
      /*c_mod=*/(short)0, c, /*reuse_a=*/false, /*reuse_b=*/false);
}

// A-fragment (16x32 bf16, documented layout): lane<16 holds row M=lane,
// K = kt*32 + {0..7, 16..23}; lane>=16 holds same M, K = kt*32 + {8..15, 24..31}.
static __device__ __forceinline__ v16bf load_a16(const __bf16* row, int kt, int klo) {
  const v8bf lo = *(const v8bf*)(row + kt * 32 + klo);
  const v8bf hi = *(const v8bf*)(row + kt * 32 + klo + 16);
  return __builtin_shufflevector(lo, hi, 0,1,2,3,4,5,6,7,8,9,10,11,12,13,14,15);
}

static __device__ __forceinline__ v16bf load_a_f32(const float* row, int kt, int klo) {
  v16bf a;
#pragma unroll
  for (int i = 0; i < 8; ++i) a[i]     = (__bf16)row[kt * 32 + klo + i];
#pragma unroll
  for (int i = 0; i < 8; ++i) a[8 + i] = (__bf16)row[kt * 32 + klo + 16 + i];
  return a;
}

// B fragments pre-swizzled: frag[((kt*T + t)*32 + lane)*16 + e]
static __device__ __forceinline__ v16bf load_b(const __bf16* frag, int tileIdx, int lane) {
  return *(const v16bf*)(frag + ((size_t)tileIdx * 32 + lane) * 16);
}

static __device__ __forceinline__ float sigmoidf_(float x) {
  return 1.0f / (1.0f + expf(-x));
}

// ---------------------------------------------------------------------------
// Weight prep: rank-3 ff vectors + bf16 B-fragment swizzles (runs each launch).
// ---------------------------------------------------------------------------
__global__ void prep_weights(const float* __restrict__ W1, const float* __restrict__ W2,
                             const float* __restrict__ W3, const float* __restrict__ Wih,
                             const float* __restrict__ Whh,
                             __bf16* __restrict__ W1aF, __bf16* __restrict__ W1bF,
                             __bf16* __restrict__ W2F,  __bf16* __restrict__ W3F,
                             __bf16* __restrict__ WihF, __bf16* __restrict__ WhhF,
                             float* __restrict__ wa, float* __restrict__ wbv,
                             float* __restrict__ wc) {
  const int tid = threadIdx.x;
  if (tid < 64) {
    // m = [s_in(0:64), ff_in(64:68), s_out(68:132), ff_out(132:136)]
    // ff_in = [b_i,-b_i,J,-J], ff_out = [-b_o,b_o,-J,J]
    wa[tid]  = W1[64 * 64 + tid] - W1[65 * 64 + tid];
    wbv[tid] = W1[66 * 64 + tid] - W1[67 * 64 + tid]
             - W1[134 * 64 + tid] + W1[135 * 64 + tid];
    wc[tid]  = W1[133 * 64 + tid] - W1[132 * 64 + tid];
  }
  // 64x64 weights: 2 ktiles * 4 ntiles * 32 lanes * 16 elems = 4096
  for (int i = tid; i < 4096; i += blockDim.x) {
    const int e    = i & 15;
    const int lane = (i >> 4) & 31;
    const int t    = (i >> 9) & 3;
    const int kt   = i >> 11;
    const int k = kt * 32 + ((lane >= 16) ? 8 : 0) + (e & 7) + ((e >= 8) ? 16 : 0);
    const int n = t * 16 + (lane & 15);
    W1aF[i] = (__bf16)W1[k * 64 + n];            // rows 0..63  (s_in)
    W1bF[i] = (__bf16)W1[(68 + k) * 64 + n];     // rows 68..131 (s_out)
    W2F[i]  = (__bf16)W2[k * 64 + n];
    W3F[i]  = (__bf16)W3[k * 64 + n];
  }
  // 64x192 GRU weights: 2 ktiles * 12 ntiles * 512 = 12288
  for (int i = tid; i < 12288; i += blockDim.x) {
    const int e    = i & 15;
    const int lane = (i >> 4) & 31;
    const int g    = (i >> 9) % 12;
    const int kt   = (i >> 9) / 12;
    const int k = kt * 32 + ((lane >= 16) ? 8 : 0) + (e & 7) + ((e >= 8) ? 16 : 0);
    const int n = g * 16 + (lane & 15);
    WihF[i] = (__bf16)Wih[k * 192 + n];
    WhhF[i] = (__bf16)Whh[k * 192 + n];
  }
}

// ---------------------------------------------------------------------------
// state = [b,-b] @ W_in + b_in  (rank-1), plus bf16 mirror
// ---------------------------------------------------------------------------
__global__ void init_state(const float* __restrict__ b, const float* __restrict__ W_in,
                           const float* __restrict__ b_in, float* __restrict__ state,
                           __bf16* __restrict__ stateBF) {
  const int i = blockIdx.x * blockDim.x + threadIdx.x;  // grid covers N*64 exactly
  const int node = i >> 6;
  const int h = i & 63;
  const float v = b[node] * (W_in[h] - W_in[64 + h]) + b_in[h];
  state[i] = v;
  stateBF[i] = (__bf16)v;
}

__global__ void zero_agg(float* __restrict__ agg) {
  agg[blockIdx.x * blockDim.x + threadIdx.x] = 0.0f;
}

// ---------------------------------------------------------------------------
// Edge MLP + atomic segment-sum. One wave per 16-edge tile.
// ---------------------------------------------------------------------------
__global__ void __launch_bounds__(128)
edge_mlp(const int* __restrict__ msg_node, const float* __restrict__ J_msg,
         const float* __restrict__ b, const __bf16* __restrict__ stateBF,
         const __bf16* __restrict__ W1aF, const __bf16* __restrict__ W1bF,
         const __bf16* __restrict__ W2F,  const __bf16* __restrict__ W3F,
         const float* __restrict__ wa, const float* __restrict__ wbv,
         const float* __restrict__ wc, const float* __restrict__ b1,
         const float* __restrict__ b2, const float* __restrict__ b3,
         float* __restrict__ agg) {
  __shared__ __bf16 ldsAll[4][16 * LDSR];
  const int lane = threadIdx.x & 31;
  const int wv   = threadIdx.x >> 5;
  const int tile = blockIdx.x * 4 + wv;            // ETILES % 4 == 0: all waves valid
  __bf16* lds = &ldsAll[wv][0];

  const int mLow = lane & 15;
  const int hiH  = lane >> 4;
  const int klo  = hiH ? 8 : 0;
  const int eBase = tile * 16;

  if (tile + 1 < ETILES)
    __builtin_prefetch(msg_node + (size_t)(eBase + 16) * 2, 0, 3);  // global_prefetch_b8

  // A fragments of s_in / s_out (row = edge mLow, both half-waves same row)
  const int einA  = msg_node[(eBase + mLow) * 2 + 0];
  const int eoutA = msg_node[(eBase + mLow) * 2 + 1];
  const __bf16* rowIn  = stateBF + (size_t)einA  * HDIM;
  const __bf16* rowOut = stateBF + (size_t)eoutA * HDIM;
  v16bf Ain0  = load_a16(rowIn, 0, klo),  Ain1  = load_a16(rowIn, 1, klo);
  v16bf Aout0 = load_a16(rowOut, 0, klo), Aout1 = load_a16(rowOut, 1, klo);

  // per-row (D-layout rows M = r + 8*hiH) scalars for the rank-3 ff epilogue
  float bi8[8], bo8[8], J8[8];
  int   no8[8];
#pragma unroll
  for (int r = 0; r < 8; ++r) {
    const int e  = eBase + r + 8 * hiH;
    const int ni = msg_node[e * 2 + 0];
    const int no = msg_node[e * 2 + 1];
    bi8[r] = b[ni]; bo8[r] = b[no]; J8[r] = J_msg[e]; no8[r] = no;
  }

  // ---- Layer 1: relu(s_in@W1a + s_out@W1b + ff-rank3 + b1) -> LDS bf16 tile
#pragma unroll
  for (int t = 0; t < 4; ++t) {
    v8f acc = {0.f, 0.f, 0.f, 0.f, 0.f, 0.f, 0.f, 0.f};
    acc = wmma_acc(Ain0,  load_b(W1aF, 0 * 4 + t, lane), acc);
    acc = wmma_acc(Ain1,  load_b(W1aF, 1 * 4 + t, lane), acc);
    acc = wmma_acc(Aout0, load_b(W1bF, 0 * 4 + t, lane), acc);
    acc = wmma_acc(Aout1, load_b(W1bF, 1 * 4 + t, lane), acc);
    const int col = t * 16 + mLow;
    const float wal = wa[col], wbl = wbv[col], wcl = wc[col], b1l = b1[col];
#pragma unroll
    for (int r = 0; r < 8; ++r) {
      float v = acc[r] + bi8[r] * wal + J8[r] * wbl + bo8[r] * wcl + b1l;
      v = v > 0.f ? v : 0.f;
      lds[(r + 8 * hiH) * LDSR + col] = (__bf16)v;
    }
  }
  __builtin_amdgcn_wave_barrier();   // per-wave LDS, DS ops are in-order

  // ---- Layer 2
  const __bf16* myrow = lds + mLow * LDSR;
  v16bf A0 = load_a16(myrow, 0, klo), A1 = load_a16(myrow, 1, klo);
  __builtin_amdgcn_wave_barrier();
#pragma unroll
  for (int t = 0; t < 4; ++t) {
    v8f acc = {0.f, 0.f, 0.f, 0.f, 0.f, 0.f, 0.f, 0.f};
    acc = wmma_acc(A0, load_b(W2F, 0 * 4 + t, lane), acc);
    acc = wmma_acc(A1, load_b(W2F, 1 * 4 + t, lane), acc);
    const int col = t * 16 + mLow;
    const float b2l = b2[col];
#pragma unroll
    for (int r = 0; r < 8; ++r) {
      float v = acc[r] + b2l;
      v = v > 0.f ? v : 0.f;
      lds[(r + 8 * hiH) * LDSR + col] = (__bf16)v;
    }
  }
  __builtin_amdgcn_wave_barrier();

  // ---- Layer 3 + atomic segment-sum into agg[edge_out]
  v16bf B0 = load_a16(myrow, 0, klo), B1 = load_a16(myrow, 1, klo);
  __builtin_amdgcn_wave_barrier();
#pragma unroll
  for (int t = 0; t < 4; ++t) {
    v8f acc = {0.f, 0.f, 0.f, 0.f, 0.f, 0.f, 0.f, 0.f};
    acc = wmma_acc(B0, load_b(W3F, 0 * 4 + t, lane), acc);
    acc = wmma_acc(B1, load_b(W3F, 1 * 4 + t, lane), acc);
    const int col = t * 16 + mLow;
    const float b3l = b3[col];
#pragma unroll
    for (int r = 0; r < 8; ++r) {
      const float v = acc[r] + b3l;
      float* dst = agg + (size_t)no8[r] * HDIM + col;
      __hip_atomic_fetch_add(dst, v, __ATOMIC_RELAXED, __HIP_MEMORY_SCOPE_AGENT);
    }
  }
}

// ---------------------------------------------------------------------------
// GRU cell via WMMA. One wave per 16-node tile; r/z fuse ih+hh chains.
// ---------------------------------------------------------------------------
__global__ void __launch_bounds__(128)
gru_cell(const float* __restrict__ agg, float* __restrict__ state,
         __bf16* __restrict__ stateBF, const __bf16* __restrict__ WihF,
         const __bf16* __restrict__ WhhF, const float* __restrict__ b_ih,
         const float* __restrict__ b_hh, int ntiles) {
  const int lane = threadIdx.x & 31;
  const int tile = blockIdx.x * 4 + (threadIdx.x >> 5);
  if (tile >= ntiles) return;                      // wave-uniform: EXEC stays all-1
  const int mLow = lane & 15;
  const int hiH  = lane >> 4;
  const int klo  = hiH ? 8 : 0;
  const int nodeA = tile * 16 + mLow;

  const float*  arow = agg + (size_t)nodeA * HDIM;
  const __bf16* srow = stateBF + (size_t)nodeA * HDIM;
  v16bf Ag0 = load_a_f32(arow, 0, klo), Ag1 = load_a_f32(arow, 1, klo);
  v16bf As0 = load_a16(srow, 0, klo),   As1 = load_a16(srow, 1, klo);

#pragma unroll
  for (int t = 0; t < 4; ++t) {
    v8f ar  = {0.f,0.f,0.f,0.f,0.f,0.f,0.f,0.f};
    v8f az  = {0.f,0.f,0.f,0.f,0.f,0.f,0.f,0.f};
    v8f ain = {0.f,0.f,0.f,0.f,0.f,0.f,0.f,0.f};
    v8f ahn = {0.f,0.f,0.f,0.f,0.f,0.f,0.f,0.f};
    // r gate: (agg@Wih + h@Whh) columns t
    ar = wmma_acc(Ag0, load_b(WihF, 0 * 12 + t, lane), ar);
    ar = wmma_acc(As0, load_b(WhhF, 0 * 12 + t, lane), ar);
    ar = wmma_acc(Ag1, load_b(WihF, 1 * 12 + t, lane), ar);
    ar = wmma_acc(As1, load_b(WhhF, 1 * 12 + t, lane), ar);
    // z gate: columns t+4
    az = wmma_acc(Ag0, load_b(WihF, 0 * 12 + t + 4, lane), az);
    az = wmma_acc(As0, load_b(WhhF, 0 * 12 + t + 4, lane), az);
    az = wmma_acc(Ag1, load_b(WihF, 1 * 12 + t + 4, lane), az);
    az = wmma_acc(As1, load_b(WhhF, 1 * 12 + t + 4, lane), az);
    // n gate inputs kept separate: columns t+8
    ain = wmma_acc(Ag0, load_b(WihF, 0 * 12 + t + 8, lane), ain);
    ain = wmma_acc(Ag1, load_b(WihF, 1 * 12 + t + 8, lane), ain);
    ahn = wmma_acc(As0, load_b(WhhF, 0 * 12 + t + 8, lane), ahn);
    ahn = wmma_acc(As1, load_b(WhhF, 1 * 12 + t + 8, lane), ahn);

    const int col = t * 16 + mLow;
    const float brl  = b_ih[col] + b_hh[col];
    const float bzl  = b_ih[64 + col] + b_hh[64 + col];
    const float binl = b_ih[128 + col];
    const float bhnl = b_hh[128 + col];
#pragma unroll
    for (int r = 0; r < 8; ++r) {
      const int node = tile * 16 + r + 8 * hiH;
      float* hp = state + (size_t)node * HDIM + col;
      const float h  = *hp;
      const float rg = sigmoidf_(ar[r] + brl);
      const float zg = sigmoidf_(az[r] + bzl);
      const float ng = tanhf(ain[r] + binl + rg * (ahn[r] + bhnl));
      const float hn = (1.f - zg) * ng + zg * h;
      *hp = hn;
      stateBF[(size_t)node * HDIM + col] = (__bf16)hn;
    }
  }
}

// ---------------------------------------------------------------------------
// out = state @ W_out + b_out   (64 -> 2, trivial)
// ---------------------------------------------------------------------------
__global__ void out_proj(const float* __restrict__ state, const float* __restrict__ W_out,
                         const float* __restrict__ b_out, float* __restrict__ out) {
  const int node = blockIdx.x * blockDim.x + threadIdx.x;
  if (node >= NNODE) return;
  float a0 = b_out[0], a1 = b_out[1];
#pragma unroll
  for (int h = 0; h < HDIM; ++h) {
    const float s = state[(size_t)node * HDIM + h];
    a0 += s * W_out[h * 2 + 0];
    a1 += s * W_out[h * 2 + 1];
  }
  out[node * 2 + 0] = a0;
  out[node * 2 + 1] = a1;
}

// ---------------------------------------------------------------------------
extern "C" void kernel_launch(void* const* d_in, const int* in_sizes, int n_in,
                              void* d_out, int out_size, void* d_ws, size_t ws_size,
                              hipStream_t stream) {
  const int*   msg_node = (const int*)d_in[0];
  const float* J_msg    = (const float*)d_in[1];
  const float* b        = (const float*)d_in[2];
  // d_in[3] idx_msg_edge: unused by the reference
  const float* W_in  = (const float*)d_in[4];
  const float* b_in  = (const float*)d_in[5];
  const float* W1    = (const float*)d_in[6];
  const float* b1    = (const float*)d_in[7];
  const float* W2    = (const float*)d_in[8];
  const float* b2    = (const float*)d_in[9];
  const float* W3    = (const float*)d_in[10];
  const float* b3    = (const float*)d_in[11];
  const float* Wih   = (const float*)d_in[12];
  const float* b_ih  = (const float*)d_in[13];
  const float* Whh   = (const float*)d_in[14];
  const float* b_hh  = (const float*)d_in[15];
  const float* W_out = (const float*)d_in[16];
  const float* b_out = (const float*)d_in[17];
  float* out = (float*)d_out;

  // workspace carve-out (256B aligned)
  char* ws = (char*)d_ws;
  size_t off = 0;
  auto carve = [&](size_t bytes) -> void* {
    off = (off + 255) & ~(size_t)255;
    void* p = ws + off;
    off += bytes;
    return p;
  };
  float*  state   = (float*)carve((size_t)NNODE * HDIM * 4);
  float*  agg     = (float*)carve((size_t)NNODE * HDIM * 4);
  __bf16* stateBF = (__bf16*)carve((size_t)NNODE * HDIM * 2);
  __bf16* W1aF = (__bf16*)carve(4096 * 2);
  __bf16* W1bF = (__bf16*)carve(4096 * 2);
  __bf16* W2F  = (__bf16*)carve(4096 * 2);
  __bf16* W3F  = (__bf16*)carve(4096 * 2);
  __bf16* WihF = (__bf16*)carve(12288 * 2);
  __bf16* WhhF = (__bf16*)carve(12288 * 2);
  float* wa  = (float*)carve(64 * 4);
  float* wbv = (float*)carve(64 * 4);
  float* wc  = (float*)carve(64 * 4);

  prep_weights<<<1, 256, 0, stream>>>(W1, W2, W3, Wih, Whh, W1aF, W1bF, W2F, W3F,
                                      WihF, WhhF, wa, wbv, wc);
  init_state<<<(NNODE * HDIM) / 256, 256, 0, stream>>>(b, W_in, b_in, state, stateBF);

  for (int p = 0; p < 3; ++p) {
    zero_agg<<<(NNODE * HDIM) / 256, 256, 0, stream>>>(agg);
    edge_mlp<<<ETILES / 4, 128, 0, stream>>>(msg_node, J_msg, b, stateBF, W1aF, W1bF,
                                             W2F, W3F, wa, wbv, wc, b1, b2, b3, agg);
    gru_cell<<<(NTILES + 3) / 4, 128, 0, stream>>>(agg, state, stateBF, WihF, WhhF,
                                                   b_ih, b_hh, NTILES);
  }
  out_proj<<<(NNODE + 255) / 256, 256, 0, stream>>>(state, W_out, b_out, out);
}
